// CrossAttention_56530359550324
// MI455X (gfx1250) — matrix operand order repaired and checked by reading, not compiled, not run on hardware
//
#include <hip/hip_runtime.h>
#include <hip/hip_bf16.h>
#include <math.h>

typedef __attribute__((ext_vector_type(16))) __bf16        v16bf;
typedef __attribute__((ext_vector_type(8)))  float         v8f;
typedef __attribute__((ext_vector_type(8)))  unsigned int  v8u;

// ---------- helpers ----------

__device__ __forceinline__ unsigned short f2bf(float f) {
    unsigned int u = __builtin_bit_cast(unsigned int, f);
    u += 0x7FFFu + ((u >> 16) & 1u);           // round-to-nearest-even
    return (unsigned short)(u >> 16);
}

__device__ __forceinline__ unsigned int pk2(float lo, float hi) {
    return (unsigned int)f2bf(lo) | ((unsigned int)f2bf(hi) << 16);
}

__device__ __forceinline__ v8f vzero8() {
    v8f z;
#pragma unroll
    for (int i = 0; i < 8; ++i) z[i] = 0.0f;
    return z;
}

// 16 contiguous bf16 (32B per lane) -> B-fragment / contiguous chunk
__device__ __forceinline__ v16bf ld_bf16_16(const unsigned short* p) {
    const uint4* q = reinterpret_cast<const uint4*>(p);
    uint4 a = q[0], b = q[1];
    v8u u; u[0]=a.x; u[1]=a.y; u[2]=a.z; u[3]=a.w; u[4]=b.x; u[5]=b.y; u[6]=b.z; u[7]=b.w;
    return __builtin_bit_cast(v16bf, u);
}

// A-matrix 16x32 bf16 pattern: per-lane contiguous chunk [2*base2, 2*base2+16)
__device__ __forceinline__ v16bf ld_bf16_apat(const unsigned short* row, int base2) {
    uint4 a = *reinterpret_cast<const uint4*>(row + 2 * base2);
    uint4 b = *reinterpret_cast<const uint4*>(row + 2 * base2 + 8);
    v8u u; u[0]=a.x; u[1]=a.y; u[2]=a.z; u[3]=a.w; u[4]=b.x; u[5]=b.y; u[6]=b.z; u[7]=b.w;
    return __builtin_bit_cast(v16bf, u);
}

__device__ __forceinline__ v8f wmma_bf16(v16bf a, v16bf b, v8f c) {
    return __builtin_amdgcn_wmma_f32_16x16x32_bf16(
        /*neg_a=*/false, a, /*neg_b=*/false, b,
        /*c_mod=*/(short)0, c, /*reuse_a=*/false, /*reuse_b=*/false);
}

// ---------- kernel 0a: bulk f32 -> bf16 conversion (row-major preserved) ----------
__global__ void cvt_bf16(const float* __restrict__ in, unsigned short* __restrict__ out, int n) {
    int t = (blockIdx.x * blockDim.x + threadIdx.x) * 4;
    if (t + 3 >= n + 1) { if (t >= n) return; }
    float4 v = *reinterpret_cast<const float4*>(in + t);
    uint2 o; o.x = pk2(v.x, v.y); o.y = pk2(v.z, v.w);
    *reinterpret_cast<uint2*>(out + t) = o;
}

// ---------- kernel 0b: transpose + f32->bf16 for weights ([K][N] -> [N][K]) ----------
__global__ void xpose_bf16(const float* __restrict__ W, unsigned short* __restrict__ Wt,
                           int K, int N) {
    int t = blockIdx.x * blockDim.x + threadIdx.x;
    if (t >= K * N) return;
    int n = t % N, k = t / N;
    Wt[(size_t)n * K + k] = f2bf(W[(size_t)k * N + n]);
}

// ---------- kernel 1: projection GEMM (Abf[bf16, MxK] @ Bt[bf16, NxK]^T) ----------
// 32x32 output tile per wave: 2 A-frags x 2 B-frags -> 4 WMMA per K-step.
// mode 0: write q [B,H,N,64] bf16, pre-scaled by 1/8
// mode 1: cols < 1024 -> k [B,H,M,64]; cols >= 1024 -> v transposed [B,H,64,M]
__device__ __forceinline__ void store_proj(int mode, float scale, int row, int col, float val,
                                           unsigned short* out_q, unsigned short* out_k,
                                           unsigned short* out_vt) {
    int bb = row >> 10, rr = row & 1023;
    if (mode == 0) {
        int h = col >> 6, d = col & 63;
        out_q[(((size_t)(bb * 16 + h)) * 1024 + rr) * 64 + d] = f2bf(val * scale);
    } else if (col < 1024) {
        int h = col >> 6, d = col & 63;
        out_k[(((size_t)(bb * 16 + h)) * 1024 + rr) * 64 + d] = f2bf(val);
    } else {
        int c = col - 1024;
        int h = c >> 6, d = c & 63;
        out_vt[(((size_t)(bb * 16 + h)) * 64 + d) * 1024 + rr] = f2bf(val);
    }
}

__global__ __launch_bounds__(256) void proj_gemm(
    const unsigned short* __restrict__ Abf, const unsigned short* __restrict__ Bt,
    int K, int N, int mode, float scale,
    unsigned short* __restrict__ out_q, unsigned short* __restrict__ out_k,
    unsigned short* __restrict__ out_vt)
{
    const int lane  = threadIdx.x & 31;
    const int warp  = threadIdx.x >> 5;
    const int tilesN = N >> 5;                     // 32-wide tiles
    const int wid = blockIdx.x * 8 + warp;
    const int tm = wid / tilesN, tn = wid % tilesN;
    const int li    = lane & 15;
    const int base2 = (lane >> 4) * 8;
    const int klo   = (lane >> 4) * 16;

    const unsigned short* arow0 = Abf + (size_t)(tm * 32 + li) * K;
    const unsigned short* arow1 = arow0 + (size_t)16 * K;
    const unsigned short* brow0 = Bt + (size_t)(tn * 32 + li) * K + klo;
    const unsigned short* brow1 = brow0 + (size_t)16 * K;

    v8f acc00 = vzero8(), acc01 = vzero8(), acc10 = vzero8(), acc11 = vzero8();
#pragma unroll 2
    for (int k0 = 0; k0 < K; k0 += 32) {
        v16bf a0 = ld_bf16_apat(arow0 + k0, base2);
        v16bf a1 = ld_bf16_apat(arow1 + k0, base2);
        v16bf b0 = ld_bf16_16(brow0 + k0);
        v16bf b1 = ld_bf16_16(brow1 + k0);
        acc00 = wmma_bf16(a0, b0, acc00);
        acc01 = wmma_bf16(a0, b1, acc01);
        acc10 = wmma_bf16(a1, b0, acc10);
        acc11 = wmma_bf16(a1, b1, acc11);
    }

#pragma unroll
    for (int e = 0; e < 8; ++e) {
        int r0 = tm * 32 + e + ((lane >> 4) << 3);
        int c0 = tn * 32 + li;
        store_proj(mode, scale, r0,      c0,      acc00[e], out_q, out_k, out_vt);
        store_proj(mode, scale, r0,      c0 + 16, acc01[e], out_q, out_k, out_vt);
        store_proj(mode, scale, r0 + 16, c0,      acc10[e], out_q, out_k, out_vt);
        store_proj(mode, scale, r0 + 16, c0 + 16, acc11[e], out_q, out_k, out_vt);
    }
}

// ---------- kernel 2: fused attention + talking heads + LN ----------
// grid 256 (b * 64 i-tiles), block 512 (16 waves, one wave per head)
__global__ __launch_bounds__(512) void attn_kernel(
    const unsigned short* __restrict__ qb, const unsigned short* __restrict__ kb,
    const unsigned short* __restrict__ vtb, const float* __restrict__ Wt,
    const float* __restrict__ gamma, const float* __restrict__ beta,
    float* __restrict__ out)
{
    __shared__ float          p_lds[16 * 32 * 17];   // [h][j(32)][i(16)+1 pad]
    __shared__ unsigned short wl_lds[16 * 16 * 40];  // [g][i][j(32)+8 pad] bf16
    __shared__ float          wt_s[256];
    __shared__ float          gam_s[16], bet_s[16];

    const int tid  = threadIdx.x;
    const int lane = tid & 31;
    const int h    = tid >> 5;                 // head for this wave
    const int b     = blockIdx.x >> 6;
    const int ibase = (blockIdx.x & 63) * 16;

    const int li    = lane & 15;
    const int base2 = (lane >> 4) * 8;
    const int klo   = (lane >> 4) * 16;

    if (tid < 256) wt_s[tid] = Wt[tid];
    if (tid < 16) { gam_s[tid] = gamma[tid]; bet_s[tid] = beta[tid]; }

    // resident Q B-fragments (B operand of S^T = K @ Q^T), d = 64 -> 2 frags
    const unsigned short* qrow = qb + (((size_t)(b * 16 + h)) * 1024 + ibase + li) * 64;
    const v16bf qf0 = ld_bf16_16(qrow + klo);
    const v16bf qf1 = ld_bf16_16(qrow + 32 + klo);

    const unsigned short* kbaseP = kb  + ((size_t)(b * 16 + h)) * 1024 * 64;
    const unsigned short* vbaseP = vtb + ((size_t)(b * 16 + h)) * 64 * 1024;

    // ---- pass 1: online softmax stats (per-lane, combined across lane^16) ----
    float m = -3.0e38f, s = 0.0f;
    for (int step = 0; step < 32; ++step) {
        const int jb = step * 32;
#pragma unroll
        for (int t = 0; t < 2; ++t) {
            const unsigned short* krow = kbaseP + (size_t)(jb + t * 16 + li) * 64;
            v16bf a0 = ld_bf16_apat(krow, base2);
            v16bf a1 = ld_bf16_apat(krow + 32, base2);
            v8f c = vzero8();
            c = wmma_bf16(a0, qf0, c);
            c = wmma_bf16(a1, qf1, c);
            float tmax = c[0];
#pragma unroll
            for (int e = 1; e < 8; ++e) tmax = fmaxf(tmax, c[e]);
            float nm = fmaxf(m, tmax);
            float ls = 0.0f;
#pragma unroll
            for (int e = 0; e < 8; ++e) ls += __expf(c[e] - nm);
            s = s * __expf(m - nm) + ls;
            m = nm;
        }
    }
    {
        float mo = __shfl_xor(m, 16, 32);
        float so = __shfl_xor(s, 16, 32);
        float M  = fmaxf(m, mo);
        s = s * __expf(m - M) + so * __expf(mo - M);
        m = M;
    }
    const float invs = 1.0f / s;

    __syncthreads();   // wt_s/gam_s/bet_s visible

    v8f acc[4];
#pragma unroll
    for (int t = 0; t < 4; ++t) acc[t] = vzero8();

    // ---- pass 2: recompute, normalize, mix heads + LN through LDS, @ V ----
    for (int step = 0; step < 32; ++step) {
        const int jb = step * 32;
#pragma unroll
        for (int t = 0; t < 2; ++t) {
            const unsigned short* krow = kbaseP + (size_t)(jb + t * 16 + li) * 64;
            v16bf a0 = ld_bf16_apat(krow, base2);
            v16bf a1 = ld_bf16_apat(krow + 32, base2);
            v8f c = vzero8();
            c = wmma_bf16(a0, qf0, c);
            c = wmma_bf16(a1, qf1, c);
#pragma unroll
            for (int e = 0; e < 8; ++e) {
                int jj = t * 16 + e + ((lane >> 4) << 3);
                p_lds[(h * 32 + jj) * 17 + li] = __expf(c[e] - m) * invs;
            }
        }
        __syncthreads();

        // one (i,j) position per thread: 16x16 head-mix + LayerNorm over heads
        {
            const int pi = tid & 15, pj = tid >> 4;    // pj in [0,32)
            float pv[16];
#pragma unroll
            for (int hh = 0; hh < 16; ++hh) pv[hh] = p_lds[(hh * 32 + pj) * 17 + pi];
            float g[16];
            float mean = 0.0f;
#pragma unroll
            for (int gg = 0; gg < 16; ++gg) {
                float a = 0.0f;
#pragma unroll
                for (int hh = 0; hh < 16; ++hh) a += pv[hh] * wt_s[hh * 16 + gg];
                g[gg] = a; mean += a;
            }
            mean *= 0.0625f;
            float var = 0.0f;
#pragma unroll
            for (int gg = 0; gg < 16; ++gg) { float d = g[gg] - mean; var += d * d; }
            var *= 0.0625f;
            float rstd = rsqrtf(var + 1e-5f);
#pragma unroll
            for (int gg = 0; gg < 16; ++gg)
                wl_lds[(gg * 16 + pi) * 40 + pj] =
                    f2bf((g[gg] - mean) * rstd * gam_s[gg] + bet_s[gg]);
        }
        __syncthreads();

        // this wave accumulates for output head g == h: acc += wln @ V
        {
            const unsigned short* wrow = &wl_lds[(h * 16 + li) * 40];
            v16bf wa = ld_bf16_apat(wrow, base2);
#pragma unroll
            for (int tn = 0; tn < 4; ++tn) {
                const unsigned short* vrow = vbaseP + (size_t)(tn * 16 + li) * 1024 + jb + klo;
                v16bf vb = ld_bf16_16(vrow);
                acc[tn] = wmma_bf16(wa, vb, acc[tn]);
            }
        }
        __syncthreads();
    }

    // ---- store out[b, ibase+r, h*64 + col] ----
#pragma unroll
    for (int tn = 0; tn < 4; ++tn) {
#pragma unroll
        for (int e = 0; e < 8; ++e) {
            int r   = e + ((lane >> 4) << 3);
            int col = h * 64 + tn * 16 + li;
            out[((size_t)(b * 1024) + ibase + r) * 1024 + col] = acc[tn][e];
        }
    }
}

// ---------- host launch ----------
extern "C" void kernel_launch(void* const* d_in, const int* in_sizes, int n_in,
                              void* d_out, int out_size, void* d_ws, size_t ws_size,
                              hipStream_t stream) {
    const float* x    = (const float*)d_in[0];
    const float* ctx  = (const float*)d_in[1];
    const float* Wq   = (const float*)d_in[2];
    const float* Wkv  = (const float*)d_in[3];
    const float* Wt   = (const float*)d_in[4];
    const float* gam  = (const float*)d_in[5];
    const float* bet  = (const float*)d_in[6];
    float* out = (float*)d_out;

    unsigned short* ws    = (unsigned short*)d_ws;
    unsigned short* xbf   = ws;                        //  4*1024*1024
    unsigned short* ctxbf = xbf   + 4 * 1024 * 1024;   //  4*1024*1024
    unsigned short* WqT   = ctxbf + 4 * 1024 * 1024;   //  1024*1024
    unsigned short* WkvT  = WqT   + 1024 * 1024;       //  2048*1024
    unsigned short* qb    = WkvT  + 2048 * 1024;       //  4*16*1024*64
    unsigned short* kb    = qb    + 4 * 16 * 1024 * 64;
    unsigned short* vtb   = kb    + 4 * 16 * 1024 * 64;

    const int nx = 4 * 1024 * 1024;
    cvt_bf16<<<nx / (256 * 4), 256, 0, stream>>>(x,   xbf,   nx);
    cvt_bf16<<<nx / (256 * 4), 256, 0, stream>>>(ctx, ctxbf, nx);

    // weights -> bf16, transposed to [out][in]
    xpose_bf16<<<(1024 * 1024 + 255) / 256, 256, 0, stream>>>(Wq, WqT, 1024, 1024);
    xpose_bf16<<<(1024 * 2048 + 255) / 256, 256, 0, stream>>>(Wkv, WkvT, 1024, 2048);

    // q = x @ Wq (scaled), kv = ctx @ Wkv   (32x32 tile per wave, 8 waves/block)
    proj_gemm<<<(4096 / 32) * (1024 / 32) / 8, 256, 0, stream>>>(
        xbf, WqT, 1024, 1024, 0, 0.125f, qb, nullptr, nullptr);
    proj_gemm<<<(4096 / 32) * (2048 / 32) / 8, 256, 0, stream>>>(
        ctxbf, WkvT, 1024, 2048, 1, 1.0f, nullptr, kb, vtb);

    // fused attention + talking heads + LN
    attn_kernel<<<4 * 64, 512, 0, stream>>>(qb, kb, vtb, Wt, gam, bet, out);
}
